// SOARALinearLayer_79912161509589
// MI455X (gfx1250) — matrix operand order repaired and checked by reading, not compiled, not run on hardware
//
#include <hip/hip_runtime.h>

#define D_IN   4096
#define D_OUT  4096
#define RANK   32
#define QBLK   64
#define SCALING 1.0f   // alpha/rank = 32/32
#define NROWS  (4 * 2048)

typedef __bf16 bf16_t;
typedef __bf16 v16bf __attribute__((ext_vector_type(16)));
typedef __bf16 v8bf  __attribute__((ext_vector_type(8)));
typedef float  v8f   __attribute__((ext_vector_type(8)));
typedef unsigned int u32x4 __attribute__((ext_vector_type(4)));

// ---------------------------------------------------------------------------
// Kernel 1: dequantize 4-bit codes (stored as int32) + per-64 scales -> bf16 W
// ---------------------------------------------------------------------------
__global__ __launch_bounds__(256)
void dequant_w_kernel(const int* __restrict__ codes,
                      const float* __restrict__ scales,
                      bf16_t* __restrict__ Wb) {
  long e0 = ((long)blockIdx.x * 256 + threadIdx.x) * 8;   // 8 elems/thread
  int o = (int)(e0 >> 12);            // / D_IN
  int i = (int)(e0 & (D_IN - 1));
  float sc = scales[o * (D_IN / QBLK) + (i >> 6)];
  v8bf outv;
#pragma unroll
  for (int t = 0; t < 8; ++t) {
    float w = ((float)codes[e0 + t] - 8.0f) * sc;
    outv[t] = (bf16_t)w;
  }
  *(v8bf*)(Wb + e0) = outv;
}

// ---------------------------------------------------------------------------
// Kernel 2: x (f32) -> bf16
// ---------------------------------------------------------------------------
__global__ __launch_bounds__(256)
void cvt_x_kernel(const float* __restrict__ x, bf16_t* __restrict__ Xb) {
  long e0 = ((long)blockIdx.x * 256 + threadIdx.x) * 8;
  v8bf outv;
#pragma unroll
  for (int t = 0; t < 8; ++t) outv[t] = (bf16_t)x[e0 + t];
  *(v8bf*)(Xb + e0) = outv;
}

// ---------------------------------------------------------------------------
// Kernel 3: out = Xb @ Wb^T + bias via v_wmma_f32_16x16x32_bf16
//   WG tile 128(M) x 256(N), BK=32. 8 waves as 2(M) x 4(N); wave tile 64x64
//   (4x4 wmma, 128 acc VGPRs -> 1.0 ds_load_b128 per wmma).
//   Global->LDS staging uses CDNA5 GLOBAL_LOAD_ASYNC_TO_LDS_B128 (ASYNCcnt),
//   double-buffered, one s_barrier per K-step.
// ---------------------------------------------------------------------------
#define BM 128
#define BN 256
#define BK 32
#define LDSK 48   // row stride (bf16): 96 bytes -> every b128 access 16B-aligned

__device__ __forceinline__ void async_ld_b128(unsigned lds_off,
                                              unsigned long long gaddr) {
  // copies 16B per lane from global gaddr to LDS at lds_off, tracked with ASYNCcnt
  asm volatile("global_load_async_to_lds_b128 %0, %1, off"
               :: "v"(lds_off), "v"(gaddr) : "memory");
}
__device__ __forceinline__ void wait_async0() {
  asm volatile("s_wait_asynccnt 0x0" ::: "memory");
}

__global__ __launch_bounds__(256)
void gemm_wmma_kernel(const bf16_t* __restrict__ Xb,
                      const bf16_t* __restrict__ Wb,
                      const float* __restrict__ bias,
                      float* __restrict__ out) {
  __shared__ __align__(16) bf16_t As[2][BM][LDSK];   // 24 KiB
  __shared__ __align__(16) bf16_t Bs[2][BN][LDSK];   // 48 KiB

  const int tid  = threadIdx.x;
  const int lane = tid & 31;
  const int wave = tid >> 5;
  const int wm   = wave >> 2;           // 0..1 -> M offset wm*64
  const int wn   = wave & 3;            // 0..3 -> N offset wn*64
  const int Mbase = blockIdx.y * BM;
  const int Nbase = blockIdx.x * BN;

  // Staging: 16B chunks. A tile = 512 chunks (2/thread), B tile = 1024 (4/thread).
  // chunk c: row = c>>2, col = (c&3)*8 (bf16 units)
  unsigned long long ga[2], gb[4];
  unsigned la[2], lb[4];
#pragma unroll
  for (int u = 0; u < 2; ++u) {
    const int c = tid + 256 * u;
    ga[u] = (unsigned long long)(uintptr_t)(Xb + (long)(Mbase + (c >> 2)) * D_IN + (c & 3) * 8);
    la[u] = (unsigned)(uintptr_t)&As[0][c >> 2][(c & 3) * 8];
  }
#pragma unroll
  for (int u = 0; u < 4; ++u) {
    const int c = tid + 256 * u;
    gb[u] = (unsigned long long)(uintptr_t)(Wb + (long)(Nbase + (c >> 2)) * D_IN + (c & 3) * 8);
    lb[u] = (unsigned)(uintptr_t)&Bs[0][c >> 2][(c & 3) * 8];
  }
  const unsigned strideA = (unsigned)sizeof(As[0]);   // bytes per buffer
  const unsigned strideB = (unsigned)sizeof(Bs[0]);

  const int m0 = lane & 15;   // row/col within 16x16 fragment
  const int kh = lane >> 4;   // K-half selector

  v8f acc[4][4] = {};

  // issue K-slab 0 into buffer 0
#pragma unroll
  for (int u = 0; u < 2; ++u) { async_ld_b128(la[u], ga[u]); ga[u] += BK * 2; }
#pragma unroll
  for (int u = 0; u < 4; ++u) { async_ld_b128(lb[u], gb[u]); gb[u] += BK * 2; }

  int buf = 0;
  for (int kt = 0; kt < D_IN / BK; ++kt) {
    wait_async0();        // this wave's async writes to LDS[buf] have landed
    __syncthreads();      // all waves' writes landed; all done reading LDS[buf^1]

    if (kt + 1 < D_IN / BK) {   // issue next slab into the other buffer
      const unsigned bo = (unsigned)(buf ^ 1);
#pragma unroll
      for (int u = 0; u < 2; ++u) { async_ld_b128(la[u] + bo * strideA, ga[u]); ga[u] += BK * 2; }
#pragma unroll
      for (int u = 0; u < 4; ++u) { async_ld_b128(lb[u] + bo * strideB, gb[u]); gb[u] += BK * 2; }
    }

    // A fragments: lane(kh=0): K 0-7 & 16-23 ; lane(kh=1): K 8-15 & 24-31
    v16bf afr[4];
#pragma unroll
    for (int mi = 0; mi < 4; ++mi) {
      const bf16_t* p = &As[buf][wm * 64 + mi * 16 + m0][0];
      v8bf lo = *(const v8bf*)(p + kh * 8);
      v8bf hi = *(const v8bf*)(p + 16 + kh * 8);
      afr[mi] = __builtin_shufflevector(lo, hi, 0,1,2,3,4,5,6,7,8,9,10,11,12,13,14,15);
    }
    // B fragments: column n = lane&15 ; lane(kh=0): K 0-7 & 16-23 ; lane(kh=1): K 8-15 & 24-31
#pragma unroll
    for (int ni = 0; ni < 4; ++ni) {
      const bf16_t* p = &Bs[buf][wn * 64 + ni * 16 + m0][0];
      v8bf lo = *(const v8bf*)(p + kh * 8);
      v8bf hi = *(const v8bf*)(p + 16 + kh * 8);
      v16bf bfr = __builtin_shufflevector(lo, hi, 0,1,2,3,4,5,6,7,8,9,10,11,12,13,14,15);
#pragma unroll
      for (int mi = 0; mi < 4; ++mi)
        acc[mi][ni] = __builtin_amdgcn_wmma_f32_16x16x32_bf16(
            false, afr[mi], false, bfr, (short)0, acc[mi][ni], false, false);
    }
    buf ^= 1;
  }

  // Epilogue: C/D layout — m = v + 8*(lane>>4), n = lane&15 ; fuse bias add
#pragma unroll
  for (int mi = 0; mi < 4; ++mi)
#pragma unroll
    for (int ni = 0; ni < 4; ++ni) {
      const int n    = Nbase + wn * 64 + ni * 16 + (lane & 15);
      const int mrow = Mbase + wm * 64 + mi * 16 + (lane >> 4) * 8;
      const float bn = bias[n];
#pragma unroll
      for (int v = 0; v < 8; ++v)
        out[(long)(mrow + v) * D_OUT + n] = acc[mi][ni][v] + bn;
    }
}

// ---------------------------------------------------------------------------
// Kernel 4: fused LoRA path, one workgroup per (b,s) row, accumulates into out
//   lo = butterfly_u( (butterfly_v(x_row) @ A.T) @ B.T ); out += SCALING*lo
// ---------------------------------------------------------------------------
__device__ __forceinline__
void butterfly_lds(float* xr, const float* __restrict__ th, int tid) {
  // levels j = 11..0  (k = 4096>>j, smallest k first, matching the reference)
  for (int j = 11; j >= 0; --j) {
    const int hbits = 11 - j;
    const int half  = 1 << hbits;
    for (int t = tid; t < D_IN / 2; t += 256) {
      const int i = t & (half - 1);
      const int p = ((t >> hbits) << (hbits + 1)) | i;
      const int q = p + half;
      float s, c;
      __sincosf(th[j * (D_IN / 2) + t], &s, &c);
      const float xp = xr[p], xq = xr[q];
      xr[p] = c * xp - s * xq;
      xr[q] = s * xp + c * xq;
    }
    __syncthreads();
  }
}

__global__ __launch_bounds__(256)
void lora_butterfly_kernel(const float* __restrict__ x,
                           const float* __restrict__ thv,
                           const float* __restrict__ thu,
                           const float* __restrict__ lora_A,
                           const float* __restrict__ lora_B,
                           float* __restrict__ out) {
  __shared__ float xr[D_IN];          // 16 KiB row buffer (reused for lo)
  __shared__ float red[RANK][8];
  __shared__ float hbuf[RANK];

  const int  tid = threadIdx.x;
  const long row = blockIdx.x;

  for (int i = tid; i < D_IN; i += 256) xr[i] = x[row * D_IN + i];
  __syncthreads();

  butterfly_lds(xr, thv, tid);        // x_rot in LDS

  // h = x_rot @ lora_A.T  (rank 32)
  float acc[RANK];
#pragma unroll
  for (int r = 0; r < RANK; ++r) acc[r] = 0.0f;
  for (int i = tid; i < D_IN; i += 256) {
    const float v = xr[i];
#pragma unroll
    for (int r = 0; r < RANK; ++r) acc[r] += v * lora_A[r * D_IN + i];
  }
  // wave32 tree reduce, then cross-wave via LDS
#pragma unroll
  for (int r = 0; r < RANK; ++r)
#pragma unroll
    for (int off = 16; off > 0; off >>= 1)
      acc[r] += __shfl_down(acc[r], off, 32);
  if ((tid & 31) == 0)
#pragma unroll
    for (int r = 0; r < RANK; ++r) red[r][tid >> 5] = acc[r];
  __syncthreads();
  if (tid < RANK) {
    float s = 0.0f;
#pragma unroll
    for (int w = 0; w < 8; ++w) s += red[tid][w];
    hbuf[tid] = s;
  }
  __syncthreads();

  float hr[RANK];
#pragma unroll
  for (int r = 0; r < RANK; ++r) hr[r] = hbuf[r];
  __syncthreads();

  // lo = h @ lora_B.T   (reuse xr)
  for (int o = tid; o < D_OUT; o += 256) {
    float s = 0.0f;
#pragma unroll
    for (int r = 0; r < RANK; ++r) s += hr[r] * lora_B[o * RANK + r];
    xr[o] = s;
  }
  __syncthreads();

  butterfly_lds(xr, thu, tid);        // butterfly_u(lo)

  for (int o = tid; o < D_OUT; o += 256)
    out[row * D_OUT + o] += SCALING * xr[o];
}

// ---------------------------------------------------------------------------
extern "C" void kernel_launch(void* const* d_in, const int* in_sizes, int n_in,
                              void* d_out, int out_size, void* d_ws, size_t ws_size,
                              hipStream_t stream) {
  const float* x        = (const float*)d_in[0];
  const int*   w_codes  = (const int*)  d_in[1];
  const float* w_scales = (const float*)d_in[2];
  const float* thv      = (const float*)d_in[3];
  const float* thu      = (const float*)d_in[4];
  const float* lora_A   = (const float*)d_in[5];
  const float* lora_B   = (const float*)d_in[6];
  const float* bias     = (const float*)d_in[7];
  float* out = (float*)d_out;

  char* ws = (char*)d_ws;
  bf16_t* Wb = (bf16_t*)ws;                                   // 32 MiB
  bf16_t* Xb = (bf16_t*)(ws + (size_t)D_OUT * D_IN * 2);      // 64 MiB

  // 1) W -> bf16
  dequant_w_kernel<<<((long)D_OUT * D_IN) / (8 * 256), 256, 0, stream>>>(
      w_codes, w_scales, Wb);
  // 2) x -> bf16
  cvt_x_kernel<<<((long)NROWS * D_IN) / (8 * 256), 256, 0, stream>>>(x, Xb);
  // 3) base GEMM + bias (writes every element of out)
  dim3 grid(D_OUT / BN, NROWS / BM);
  gemm_wmma_kernel<<<grid, 256, 0, stream>>>(Xb, Wb, bias, out);
  // 4) fused LoRA/butterfly accumulate (same stream => ordered after 3)
  lora_butterfly_kernel<<<NROWS, 256, 0, stream>>>(x, thv, thu, lora_A, lora_B, out);
}
